// AttentionalPropagation_23897198035157
// MI455X (gfx1250) — matrix operand order repaired and checked by reading, not compiled, not run on hardware
//
#include <hip/hip_runtime.h>
#include <cstdint>

typedef __attribute__((ext_vector_type(16))) _Float16 v16h;
typedef __attribute__((ext_vector_type(8)))  _Float16 v8h;
typedef __attribute__((ext_vector_type(8)))  float    v8f;

#define B_   4
#define D_   256
#define H_   4
#define N_   2048
#define HN_  8192      // H_*N_
#define M_   32768     // B_*HN_

// ---------------------------------------------------------------- CDNA5 async copy
// global -> LDS DMA, tracked by ASYNCcnt (ISA 15.18.3 op 98). Per-lane: 16B.
__device__ __forceinline__ void async_load_b128(const _Float16* lds_dst, const _Float16* gsrc) {
  const uint32_t off = (uint32_t)(uintptr_t)lds_dst;          // low 32 bits = LDS offset
  const uint64_t ga  = (uint64_t)(uintptr_t)gsrc;
  asm volatile("global_load_async_to_lds_b128 %0, %1, off"
               :: "v"(off), "v"(ga) : "memory");
}
__device__ __forceinline__ void wait_async0() {
  asm volatile("s_wait_asynccnt 0" ::: "memory");
}

// ---------------------------------------------------------------- fragments
__device__ __forceinline__ v16h cat16(v8h lo, v8h hi) {
  return __builtin_shufflevector(lo, hi, 0,1,2,3,4,5,6,7,8,9,10,11,12,13,14,15);
}

// A operand (16x32 f16, MxK) from row-major [row][K] storage (ISA 7.12.2)
__device__ __forceinline__ v16h load_a_frag(const _Float16* base, int stride, int lane) {
  const int row = lane & 15, hl = lane >> 4;
  const _Float16* p = base + row * stride + 8 * hl;
  v8h lo = *(const v8h*)p;
  v8h hi = *(const v8h*)(p + 16);
  return cat16(lo, hi);
}

// B operand (32x16 f16, KxN) staged as [N][K] row-major
__device__ __forceinline__ v16h load_b_frag(const _Float16* base, int stride, int lane) {
  const int row = lane & 15, hl = lane >> 4;
  const _Float16* p = base + row * stride + 8 * hl;
  v8h lo = *(const v8h*)p;
  v8h hi = *(const v8h*)(p + 16);
  return cat16(lo, hi);
}

__device__ __forceinline__ v8f wmma16(v16h a, v16h b, v8f c) {
  return __builtin_amdgcn_wmma_f32_16x16x32_f16(false, a, false, b, (short)0, c, false, false);
}

__device__ __forceinline__ v8f zero8() {
  v8f z = {0.f,0.f,0.f,0.f,0.f,0.f,0.f,0.f};
  return z;
}

// ---------------------------------------------------------------- converts
__global__ __launch_bounds__(256) void cvt_f32_f16(const float* __restrict__ in,
                                                   _Float16* __restrict__ out, int n, float s) {
  int i = blockIdx.x * 256 + threadIdx.x;
  if (i < n) out[i] = (_Float16)(in[i] * s);
}

__global__ __launch_bounds__(256) void scale_f32(const float* __restrict__ in,
                                                 float* __restrict__ out, int n, float s) {
  int i = blockIdx.x * 256 + threadIdx.x;
  if (i < n) out[i] = in[i] * s;
}

// [b][c][hn] f32 -> activation layout [b*hn][c] f16, LDS-tiled transpose
__global__ __launch_bounds__(256) void transpose_convert(const float* __restrict__ in,
                                                         _Float16* __restrict__ out) {
  __shared__ float tile[32][33];
  const int b = blockIdx.z, c0 = blockIdx.y * 32, n0 = blockIdx.x * 32;
  const int tx = threadIdx.x, ty = threadIdx.y;
  const float* src = in + ((size_t)b * D_ + c0) * HN_ + n0;
  #pragma unroll
  for (int i = 0; i < 32; i += 8)
    tile[ty + i][tx] = src[(size_t)(ty + i) * HN_ + tx];
  __syncthreads();
  _Float16* dst = out + ((size_t)b * HN_ + n0) * D_ + c0;
  #pragma unroll
  for (int i = 0; i < 32; i += 8)
    dst[(size_t)(ty + i) * D_ + tx] = (_Float16)tile[tx][ty + i];
}

// ---------------------------------------------------------------- GEMM
// C[m][o] = sum_c A[m][c] * W[o][c] + bias[o]
// Double-buffered LDS, async global->LDS staging, one barrier per K-step.
// MODE 0: f16 act-layout store   MODE 1: f32 act-layout store
// MODE 2: f32 channel-major [b][o][hn] store via LDS re-transpose
template<int MODE>
__global__ __launch_bounds__(256) void gemm_kernel(const _Float16* __restrict__ A,
                                                   const _Float16* __restrict__ W,
                                                   const float* __restrict__ bias,
                                                   void* __restrict__ Cout,
                                                   int K, int ldc, int coloff) {
  __shared__ __align__(32) _Float16 Asm[2][128 * 32];
  __shared__ __align__(32) _Float16 Bsm[2][64 * 32];
  __shared__ float Osm[(MODE == 2) ? 64 * 132 : 1];

  const int tid = threadIdx.x;
  const int lane = tid & 31, wid = tid >> 5;
  const int r16 = lane & 15, hl = lane >> 4;
  const int mblk = blockIdx.x * 128;
  const int nblk = blockIdx.y * 64;
  const int waveM = (wid & 3) * 32;
  const int waveN = (wid >> 2) * 32;

  const int ra = tid >> 1, ca = (tid & 1) * 16;   // A staging slot (2x16B / thread)
  const int rb = tid >> 2, cb = (tid & 3) * 8;    // B staging slot (1x16B / thread)

  auto stage = [&](int p, int k0) {
    const _Float16* ga = A + (size_t)(mblk + ra) * K + k0 + ca;
    async_load_b128(&Asm[p][ra * 32 + ca], ga);
    async_load_b128(&Asm[p][ra * 32 + ca + 8], ga + 8);
    async_load_b128(&Bsm[p][rb * 32 + cb], W + (size_t)(nblk + rb) * K + k0 + cb);
  };

  v8f acc[2][2] = { { zero8(), zero8() }, { zero8(), zero8() } };

  stage(0, 0);
  wait_async0();
  __syncthreads();

  int p = 0;
  for (int k0 = 0; k0 < K; k0 += 32) {
    const bool more = (k0 + 32) < K;
    if (more) stage(1 - p, k0 + 32);            // DMA overlaps with WMMA below

    v16h af[2], bf[2];
    #pragma unroll
    for (int mi = 0; mi < 2; ++mi)
      af[mi] = load_a_frag(&Asm[p][(waveM + mi * 16) * 32], 32, lane);
    #pragma unroll
    for (int ni = 0; ni < 2; ++ni)
      bf[ni] = load_b_frag(&Bsm[p][(waveN + ni * 16) * 32], 32, lane);
    #pragma unroll
    for (int mi = 0; mi < 2; ++mi)
      #pragma unroll
      for (int ni = 0; ni < 2; ++ni)
        acc[mi][ni] = wmma16(af[mi], bf[ni], acc[mi][ni]);

    if (more) {
      wait_async0();
      __syncthreads();
      p ^= 1;
    }
  }

  if (MODE == 0 || MODE == 1) {
    #pragma unroll
    for (int mi = 0; mi < 2; ++mi)
      #pragma unroll
      for (int ni = 0; ni < 2; ++ni) {
        const int gn = nblk + waveN + ni * 16 + r16;
        const float bv = bias[gn];
        #pragma unroll
        for (int r = 0; r < 8; ++r) {
          const size_t gm = (size_t)mblk + waveM + mi * 16 + r + 8 * hl;
          const float val = acc[mi][ni][r] + bv;
          if (MODE == 0) ((_Float16*)Cout)[gm * ldc + coloff + gn] = (_Float16)val;
          else           ((float*)Cout)[gm * ldc + coloff + gn] = val;
        }
      }
  } else {
    #pragma unroll
    for (int mi = 0; mi < 2; ++mi)
      #pragma unroll
      for (int ni = 0; ni < 2; ++ni) {
        const int ol = waveN + ni * 16 + r16;
        const float bv = bias[nblk + ol];
        #pragma unroll
        for (int r = 0; r < 8; ++r) {
          const int ml = waveM + mi * 16 + r + 8 * hl;
          Osm[ol * 132 + ml] = acc[mi][ni][r] + bv;
        }
      }
    __syncthreads();
    const int b = mblk >> 13;
    const int hn0 = mblk & (HN_ - 1);
    float* out = (float*)Cout;
    for (int i = tid; i < 64 * 128; i += 256) {
      const int o = i >> 7, m = i & 127;
      out[(size_t)b * D_ * HN_ + (size_t)(nblk + o) * HN_ + hn0 + m] = Osm[o * 132 + m];
    }
  }
}

// ---------------------------------------------------------------- flash attention
// q,k,v,msg: activation layout [bh*2048 + n][256] f16.  q pre-scaled by 1/16.
// Double-buffered K (async DMA) and V (register prefetch + LDS transpose);
// one barrier per 32-key block.
__global__ __launch_bounds__(256) void attention_kernel(const _Float16* __restrict__ q,
                                                        const _Float16* __restrict__ k,
                                                        const _Float16* __restrict__ v,
                                                        _Float16* __restrict__ msg) {
  __shared__ __align__(32) _Float16 Qs[128 * 256];     // 64 KB
  __shared__ __align__(32) _Float16 Ks[2][32 * 256];   // 2 x 16 KB
  __shared__ __align__(32) _Float16 Vt[2][256 * 40];   // 2 x 20 KB (transposed, padded)
  __shared__ __align__(32) _Float16 Ps[8][16 * 40];    // 10 KB, per-wave P tiles

  const int tid = threadIdx.x, lane = tid & 31, wid = tid >> 5;
  const int r16 = lane & 15, hl = lane >> 4;
  const size_t rowbase = (size_t)blockIdx.y * 2048;    // (b,h) base row
  const int q0 = blockIdx.x * 128;

  // async-stage all 128 query rows (16 x b128 per thread)
  for (int i = tid; i < 128 * 32; i += 256) {
    const int row = i >> 5, c = (i & 31) * 8;
    async_load_b128(Qs + row * 256 + c, q + (rowbase + q0 + row) * 256 + c);
  }
  // async-stage K block 0 (4 x b128 per thread)
  #pragma unroll
  for (int t = 0; t < 4; ++t) {
    const int i = tid + 256 * t;
    const int row = i >> 5, c = (i & 31) * 8;
    async_load_b128(&Ks[0][row * 256 + c], k + (rowbase + row) * 256 + c);
  }
  // V block 0: registers -> transposed LDS
  uint4 vr[4];
  #pragma unroll
  for (int t = 0; t < 4; ++t) {
    const int i = tid + 256 * t;
    const int key = i >> 5, d0 = (i & 31) * 8;
    vr[t] = *(const uint4*)(v + (rowbase + key) * 256 + d0);
  }
  #pragma unroll
  for (int t = 0; t < 4; ++t) {
    const int i = tid + 256 * t;
    const int key = i >> 5, d0 = (i & 31) * 8;
    _Float16 h8[8];
    *(uint4*)h8 = vr[t];
    #pragma unroll
    for (int j = 0; j < 8; ++j) Vt[0][(d0 + j) * 40 + key] = h8[j];
  }
  wait_async0();
  __syncthreads();

  v8f acc[16];
  #pragma unroll
  for (int t = 0; t < 16; ++t) acc[t] = zero8();
  float mrow[8], lrow[8];
  #pragma unroll
  for (int r = 0; r < 8; ++r) { mrow[r] = -1e30f; lrow[r] = 0.0f; }

  int p = 0;
  for (int kb = 0; kb < 2048; kb += 32) {
    const bool more = (kb + 32) < 2048;
    if (more) {
      #pragma unroll
      for (int t = 0; t < 4; ++t) {          // async K for next block
        const int i = tid + 256 * t;
        const int row = i >> 5, c = (i & 31) * 8;
        async_load_b128(&Ks[1 - p][row * 256 + c],
                        k + (rowbase + kb + 32 + row) * 256 + c);
      }
      #pragma unroll
      for (int t = 0; t < 4; ++t) {          // V next block -> regs
        const int i = tid + 256 * t;
        const int key = i >> 5, d0 = (i & 31) * 8;
        vr[t] = *(const uint4*)(v + (rowbase + kb + 32 + key) * 256 + d0);
      }
    }

    // S = (Q/16) K^T for 16 queries x 32 keys
    v8f S0 = zero8(), S1 = zero8();
    #pragma unroll
    for (int kd = 0; kd < 256; kd += 32) {
      const v16h aq = load_a_frag(Qs + (wid * 16) * 256 + kd, 256, lane);
      const v16h b0 = load_b_frag(&Ks[p][kd], 256, lane);
      const v16h b1 = load_b_frag(&Ks[p][16 * 256 + kd], 256, lane);
      S0 = wmma16(aq, b0, S0);
      S1 = wmma16(aq, b1, S1);
    }

    // online softmax (row q = r + 8*hl, key = lane%16 within each half)
    float corr[8];
    _Float16* pw = &Ps[wid][0];
    #pragma unroll
    for (int r = 0; r < 8; ++r) {
      const float s0 = S0[r], s1 = S1[r];
      float t = fmaxf(s0, s1);
      #pragma unroll
      for (int off = 1; off < 16; off <<= 1) t = fmaxf(t, __shfl_xor(t, off, 16));
      const float mn = fmaxf(mrow[r], t);
      corr[r] = __expf(mrow[r] - mn);
      const float p0 = __expf(s0 - mn), p1 = __expf(s1 - mn);
      float rs = p0 + p1;
      #pragma unroll
      for (int off = 1; off < 16; off <<= 1) rs += __shfl_xor(rs, off, 16);
      lrow[r] = lrow[r] * corr[r] + rs;
      mrow[r] = mn;
      pw[(r + 8 * hl) * 40 + r16]      = (_Float16)p0;
      pw[(r + 8 * hl) * 40 + 16 + r16] = (_Float16)p1;
    }
    #pragma unroll
    for (int t = 0; t < 16; ++t)
      #pragma unroll
      for (int r = 0; r < 8; ++r) acc[t][r] *= corr[r];

    // acc += P V   (per-wave LDS roundtrip; LDS is in-order within a wave)
    const v16h ap = load_a_frag(pw, 40, lane);
    #pragma unroll
    for (int dt = 0; dt < 16; ++dt) {
      const v16h bv = load_b_frag(&Vt[p][(dt * 16) * 40], 40, lane);
      acc[dt] = wmma16(ap, bv, acc[dt]);
    }

    if (more) {
      #pragma unroll
      for (int t = 0; t < 4; ++t) {          // scatter prefetched V into other buffer
        const int i = tid + 256 * t;
        const int key = i >> 5, d0 = (i & 31) * 8;
        _Float16 h8[8];
        *(uint4*)h8 = vr[t];
        #pragma unroll
        for (int j = 0; j < 8; ++j) Vt[1 - p][(d0 + j) * 40 + key] = h8[j];
      }
      wait_async0();
      __syncthreads();
      p ^= 1;
    }
  }

  #pragma unroll
  for (int r = 0; r < 8; ++r) lrow[r] = 1.0f / lrow[r];
  #pragma unroll
  for (int dt = 0; dt < 16; ++dt)
    #pragma unroll
    for (int r = 0; r < 8; ++r) {
      const size_t m = rowbase + q0 + wid * 16 + r + 8 * hl;
      msg[m * 256 + dt * 16 + r16] = (_Float16)(acc[dt][r] * lrow[r]);
    }
}

// ---------------------------------------------------------------- hcat left half
__global__ __launch_bounds__(256) void copy_xt_hcat(const _Float16* __restrict__ xt,
                                                    _Float16* __restrict__ hcat) {
  const int i = blockIdx.x * 256 + threadIdx.x;   // M_*32 uint4 chunks
  const int m = i >> 5, c8 = i & 31;
  *(uint4*)(hcat + (size_t)m * 512 + c8 * 8) = *(const uint4*)(xt + (size_t)m * 256 + c8 * 8);
}

// ---------------------------------------------------------------- BatchNorm
__global__ __launch_bounds__(512) void bn_partial(const float* __restrict__ h,
                                                  float* __restrict__ part) {
  const int c = threadIdx.x, blk = blockIdx.x;    // 64 blocks x 512 rows
  float s = 0.f, sq = 0.f;
  const float* p = h + (size_t)blk * 512 * 512 + c;
  for (int r = 0; r < 512; ++r) { const float v = p[(size_t)r * 512]; s += v; sq += v * v; }
  part[blk * 1024 + c] = s;
  part[blk * 1024 + 512 + c] = sq;
}

__global__ __launch_bounds__(512) void bn_finalize(const float* __restrict__ part,
                                                   const float* __restrict__ gamma,
                                                   const float* __restrict__ beta,
                                                   float* __restrict__ sshift) {
  const int c = threadIdx.x;
  float s = 0.f, sq = 0.f;
  for (int b = 0; b < 64; ++b) { s += part[b * 1024 + c]; sq += part[b * 1024 + 512 + c]; }
  const float mu  = s * (1.0f / 32768.0f);
  const float var = sq * (1.0f / 32768.0f) - mu * mu;
  const float rs  = rsqrtf(var + 1e-5f);
  const float scl = gamma[c] * rs;
  sshift[c] = scl;
  sshift[512 + c] = beta[c] - mu * scl;
}

__global__ __launch_bounds__(256) void bn_apply(const float* __restrict__ h1,
                                                const float* __restrict__ sshift,
                                                _Float16* __restrict__ h2) {
  const size_t i = (size_t)blockIdx.x * 256 + threadIdx.x;  // 16777216 elements
  const int c = (int)(i & 511);
  const float v = h1[i] * sshift[c] + sshift[512 + c];
  h2[i] = (_Float16)fmaxf(v, 0.0f);
}

// ---------------------------------------------------------------- launcher
extern "C" void kernel_launch(void* const* d_in, const int* in_sizes, int n_in,
                              void* d_out, int out_size, void* d_ws, size_t ws_size,
                              hipStream_t stream) {
  const float* x    = (const float*)d_in[0];
  const float* srcv = (const float*)d_in[1];
  const float* Wq  = (const float*)d_in[2];  const float* bq = (const float*)d_in[3];
  const float* Wk  = (const float*)d_in[4];  const float* bk = (const float*)d_in[5];
  const float* Wv  = (const float*)d_in[6];  const float* bv = (const float*)d_in[7];
  const float* Wm  = (const float*)d_in[8];  const float* bm = (const float*)d_in[9];
  const float* W1  = (const float*)d_in[10]; const float* b1 = (const float*)d_in[11];
  const float* gamma = (const float*)d_in[12];
  const float* beta  = (const float*)d_in[13];
  const float* W2  = (const float*)d_in[14]; const float* b2 = (const float*)d_in[15];
  float* out = (float*)d_out;

  char* p = (char*)d_ws;
  auto take = [&](size_t bytes) { char* r = p; p += (bytes + 255) & ~(size_t)255; return r; };

  _Float16* Wq16 = (_Float16*)take((size_t)D_ * D_ * 2);
  _Float16* Wk16 = (_Float16*)take((size_t)D_ * D_ * 2);
  _Float16* Wv16 = (_Float16*)take((size_t)D_ * D_ * 2);
  _Float16* Wm16 = (_Float16*)take((size_t)D_ * D_ * 2);
  _Float16* W116 = (_Float16*)take((size_t)512 * 512 * 2);
  _Float16* W216 = (_Float16*)take((size_t)256 * 512 * 2);
  _Float16* xt   = (_Float16*)take((size_t)M_ * D_ * 2);
  _Float16* st   = (_Float16*)take((size_t)M_ * D_ * 2);
  _Float16* q16  = (_Float16*)take((size_t)M_ * D_ * 2);
  _Float16* k16  = (_Float16*)take((size_t)M_ * D_ * 2);
  _Float16* v16  = (_Float16*)take((size_t)M_ * D_ * 2);
  _Float16* msg16= (_Float16*)take((size_t)M_ * D_ * 2);
  _Float16* hcat = (_Float16*)take((size_t)M_ * 512 * 2);
  float*    h1   = (float*)   take((size_t)M_ * 512 * 4);
  _Float16* h2   = (_Float16*)take((size_t)M_ * 512 * 2);
  float*    part = (float*)   take((size_t)64 * 1024 * 4);
  float*    ssh  = (float*)   take((size_t)1024 * 4);
  float*    bqs  = (float*)   take((size_t)D_ * 4);

  const float qscale = 0.0625f;   // 1/sqrt(256), folded into Wq/bq

  // weight converts (Wq scaled by 1/16)
  cvt_f32_f16<<<256, 256, 0, stream>>>(Wq, Wq16, D_ * D_, qscale);
  cvt_f32_f16<<<256, 256, 0, stream>>>(Wk, Wk16, D_ * D_, 1.0f);
  cvt_f32_f16<<<256, 256, 0, stream>>>(Wv, Wv16, D_ * D_, 1.0f);
  cvt_f32_f16<<<256, 256, 0, stream>>>(Wm, Wm16, D_ * D_, 1.0f);
  cvt_f32_f16<<<1024, 256, 0, stream>>>(W1, W116, 512 * 512, 1.0f);
  cvt_f32_f16<<<512, 256, 0, stream>>>(W2, W216, 256 * 512, 1.0f);
  scale_f32<<<1, 256, 0, stream>>>(bq, bqs, D_, qscale);

  // activation transpose+convert
  transpose_convert<<<dim3(HN_ / 32, D_ / 32, B_), dim3(32, 8), 0, stream>>>(x, xt);
  transpose_convert<<<dim3(HN_ / 32, D_ / 32, B_), dim3(32, 8), 0, stream>>>(srcv, st);

  // q,k,v projections (f16 act layout); q pre-scaled
  gemm_kernel<0><<<dim3(M_ / 128, 4), 256, 0, stream>>>(xt, Wq16, bqs, q16, 256, 256, 0);
  gemm_kernel<0><<<dim3(M_ / 128, 4), 256, 0, stream>>>(st, Wk16, bk, k16, 256, 256, 0);
  gemm_kernel<0><<<dim3(M_ / 128, 4), 256, 0, stream>>>(st, Wv16, bv, v16, 256, 256, 0);

  // flash attention: grid (query tiles, b*h)
  attention_kernel<<<dim3(N_ / 128, B_ * H_), 256, 0, stream>>>(q16, k16, v16, msg16);

  // message projection straight into right half of concat buffer
  gemm_kernel<0><<<dim3(M_ / 128, 4), 256, 0, stream>>>(msg16, Wm16, bm, hcat, 256, 512, 256);
  copy_xt_hcat<<<(M_ * 32) / 256, 256, 0, stream>>>(xt, hcat);

  // MLP layer 1 (f32 out for BN stats)
  gemm_kernel<1><<<dim3(M_ / 128, 8), 256, 0, stream>>>(hcat, W116, b1, h1, 512, 512, 0);

  // BatchNorm (deterministic two-stage) + ReLU + f16
  bn_partial<<<64, 512, 0, stream>>>(h1, part);
  bn_finalize<<<1, 512, 0, stream>>>(part, gamma, beta, ssh);
  bn_apply<<<(M_ * 512) / 256, 256, 0, stream>>>(h1, ssh, h2);

  // MLP layer 2 -> f32 channel-major output [b][o][hn]
  gemm_kernel<2><<<dim3(M_ / 128, 4), 256, 0, stream>>>(h2, W216, b2, out, 512, 0, 0);
}
